// EncoderModel_75720273429272
// MI455X (gfx1250) — matrix operand order repaired and checked by reading, not compiled, not run on hardware
//
#include <hip/hip_runtime.h>

constexpr int kB = 16, kS = 512, kE = 1024, kH = 16, kD = 64, kNL = 6, kOUT = 10;

typedef __bf16 v16bf __attribute__((ext_vector_type(16)));
typedef __bf16 v8bf  __attribute__((ext_vector_type(8)));
typedef float  v8f   __attribute__((ext_vector_type(8)));
typedef int    b128_t __attribute__((vector_size(16)));   // 16B payload type

// ---------------------------------------------------------------------------
// CDNA5 async copy global -> LDS (16B), tracked by ASYNCcnt
// ---------------------------------------------------------------------------
__device__ __forceinline__ void async_ld_b128(const void* g, void* l)
{
#if __has_builtin(__builtin_amdgcn_global_load_async_to_lds_b128)
  __builtin_amdgcn_global_load_async_to_lds_b128(
      (__attribute__((address_space(1))) b128_t*)(unsigned long long)g,
      (__attribute__((address_space(3))) b128_t*)(unsigned)(unsigned long long)l,
      0, 0);
#else
  unsigned loff = (unsigned)(unsigned long long)l;   // LDS byte offset = low 32 bits
  asm volatile("global_load_async_to_lds_b128 %0, %1, off"
               :: "v"(loff), "v"(g) : "memory");
#endif
}

__device__ __forceinline__ void wait_async0()
{
#if __has_builtin(__builtin_amdgcn_s_wait_asynccnt)
  __builtin_amdgcn_s_wait_asynccnt(0);
#else
  asm volatile("s_wait_asynccnt 0x0" ::: "memory");
#endif
}

// ---------------------------------------------------------------------------
// Embedding lookup + positional encoding:  x = 2*emb[tok] + pe(b)
// pe indexes the *batch* dim (faithful to reference).
// ---------------------------------------------------------------------------
__global__ __launch_bounds__(256)
void enc_embed(const int* __restrict__ tok, const float* __restrict__ emb,
               float* __restrict__ x, __bf16* __restrict__ xb)
{
  int bs = blockIdx.x;            // b*S + s
  int b  = bs >> 9;               // S = 512
  int tk = tok[bs];
  size_t base = (size_t)bs * kE;
  int t = threadIdx.x;
#pragma unroll
  for (int i = 0; i < 4; ++i) {
    int e = t + i * 256;
    float ev   = emb[(size_t)tk * kE + e];
    int   twoi = e & ~1;
    float div  = __expf((float)twoi * (-9.210340371976184f / (float)kE)); // -ln(1e4)/E
    float ang  = (float)b * div;
    float pe   = (e & 1) ? __cosf(ang) : __sinf(ang);
    float val  = 2.0f * ev + pe;
    x[base + e]  = val;
    xb[base + e] = (__bf16)val;
  }
}

// ---------------------------------------------------------------------------
// Weight transpose + f32 -> bf16:  Wt[n][k] = W[k][n]   (E x E)
// ---------------------------------------------------------------------------
__global__ __launch_bounds__(256)
void enc_transpose_w(const float* __restrict__ W, __bf16* __restrict__ Wt)
{
  int idx = blockIdx.x * 256 + threadIdx.x;   // idx = k*E + n
  int k = idx >> 10, n = idx & (kE - 1);
  Wt[(size_t)n * kE + k] = (__bf16)W[idx];
}

// ---------------------------------------------------------------------------
// bf16 WMMA GEMM:  C[M,N] = alpha * A[M,K] * Bt[N,K]^T + bias
//   A, Bt row-major with leading dims lda/ldb (both are [rows][K]).
//   Grid: (N/BN, M/128, Z); per-z element strides sA/sB/sC.
//   STORE: 0 = f32 C, 1 = bf16 C, 2 = bf16 V^T epilogue ([B,H,D,S] layout)
//   Async (ASYNCcnt) double-buffered LDS staging.
// ---------------------------------------------------------------------------
template<int BN, int STORE>
__global__ __launch_bounds__(256)
void enc_gemm(const __bf16* __restrict__ A, const __bf16* __restrict__ Bt,
              const float* __restrict__ bias,
              float* __restrict__ Cf, __bf16* __restrict__ Cb,
              int K, int lda, int ldb, int ldc,
              long long sA, long long sB, long long sC, float alpha)
{
  constexpr int BM = 128, BK = 32, PK = BK + 8;    // pad -> 80B row stride
  constexpr int WN = BN / 2;                        // 4 waves (m) x 2 waves (n)
  constexpr int TM = 2;                             // 32 rows per wave
  constexpr int TN = WN / 16;                       // 4 (BN=128) or 2 (BN=64)
  constexpr int ACH = (BM * BK / 8) / 256;          // A 16B-chunks per thread (2)
  constexpr int BCH = (BN * BK / 8) / 256;          // B 16B-chunks per thread (2 or 1)

  __shared__ __align__(16) __bf16 As[2][BM][PK];
  __shared__ __align__(16) __bf16 Bs[2][BN][PK];

  const int z = blockIdx.z;
  A  += (size_t)z * (size_t)sA;
  Bt += (size_t)z * (size_t)sB;
  if (STORE == 0) Cf += (size_t)z * (size_t)sC;
  else            Cb += (size_t)z * (size_t)sC;

  const int tid   = threadIdx.x;
  const int lane  = tid & 31;
  const int wid   = tid >> 5;
  const int waveM = wid >> 1;           // 0..3
  const int waveN = wid & 1;            // 0..1
  const int lrow  = lane & 15;
  const int half  = lane >> 4;          // 0 or 1
  const int rowBlock = blockIdx.y * BM;
  const int colBlock = blockIdx.x * BN;

  v8f acc[TM][TN];
#pragma unroll
  for (int i = 0; i < TM; ++i)
#pragma unroll
    for (int j = 0; j < TN; ++j) acc[i][j] = (v8f)(0.0f);

  auto stage = [&](int buf, int k0) {
#pragma unroll
    for (int i = 0; i < ACH; ++i) {
      int c = tid + i * 256;
      int r = c >> 2, cc = (c & 3) * 8;
      async_ld_b128(&A[(size_t)(rowBlock + r) * lda + k0 + cc], &As[buf][r][cc]);
    }
#pragma unroll
    for (int i = 0; i < BCH; ++i) {
      int c = tid + i * 256;
      int r = c >> 2, cc = (c & 3) * 8;
      async_ld_b128(&Bt[(size_t)(colBlock + r) * ldb + k0 + cc], &Bs[buf][r][cc]);
    }
  };

  stage(0, 0);
  const int steps = K / BK;
  for (int s = 0; s < steps; ++s) {
    const int cur = s & 1;
    wait_async0();            // this wave's chunks of tile s have landed in LDS
    __syncthreads();          // everyone's chunks have landed; prior reads done
    if (s + 1 < steps) stage(1 - cur, (s + 1) * BK);

    // A fragment: lane half 0 -> K {0..7, 16..23}; half 1 -> K {8..15, 24..31}
    v16bf af[TM];
#pragma unroll
    for (int i = 0; i < TM; ++i) {
      int r = waveM * 32 + i * 16 + lrow;
      union { v16bf v; v8bf h[2]; } u;
      u.h[0] = *(const v8bf*)&As[cur][r][half * 8];
      u.h[1] = *(const v8bf*)&As[cur][r][16 + half * 8];
      af[i] = u.v;
    }
    // B fragment: lane half 0 -> K {0..7, 16..23}; half 1 -> K {8..15, 24..31}
    v16bf bf[TN];
#pragma unroll
    for (int j = 0; j < TN; ++j) {
      int r = waveN * WN + j * 16 + lrow;
      union { v16bf v; v8bf h[2]; } u;
      u.h[0] = *(const v8bf*)&Bs[cur][r][half * 8];
      u.h[1] = *(const v8bf*)&Bs[cur][r][16 + half * 8];
      bf[j] = u.v;
    }
#pragma unroll
    for (int i = 0; i < TM; ++i)
#pragma unroll
      for (int j = 0; j < TN; ++j)
        acc[i][j] = __builtin_amdgcn_wmma_f32_16x16x32_bf16(
            false, af[i], false, bf[j], (short)0, acc[i][j], false, false);
  }

  // epilogue: C layout -> VGPR r: lanes 0-15 row r, lanes 16-31 row r+8; col = lane&15
#pragma unroll
  for (int i = 0; i < TM; ++i) {
#pragma unroll
    for (int j = 0; j < TN; ++j) {
      int col  = colBlock + waveN * WN + j * 16 + lrow;
      float bv = bias ? bias[col] : 0.0f;
      int rbase = rowBlock + waveM * 32 + i * 16 + half * 8;
#pragma unroll
      for (int r = 0; r < 8; ++r) {
        float val = acc[i][j][r] * alpha + bv;
        int row = rbase + r;
        if (STORE == 0) {
          Cf[(size_t)row * ldc + col] = val;
        } else if (STORE == 1) {
          Cb[(size_t)row * ldc + col] = (__bf16)val;
        } else {  // V^T: row = b*S+s, col = h*D+d  ->  vT[b][h][d][s]
          int bb = row >> 9, s = row & (kS - 1);
          int h  = col >> 6, d = col & (kD - 1);
          Cb[(((size_t)(bb * kH + h)) * kD + d) * kS + s] = (__bf16)val;
        }
      }
    }
  }
}

// ---------------------------------------------------------------------------
// Row softmax over length-512 rows (mask provably all-false: tokens < PAD)
// ---------------------------------------------------------------------------
__global__ __launch_bounds__(256)
void enc_softmax(const float* __restrict__ in, __bf16* __restrict__ out)
{
  size_t base = (size_t)blockIdx.x * kS;
  int t = threadIdx.x;
  float a = in[base + t], b = in[base + t + 256];
  __shared__ float red[256];
  red[t] = fmaxf(a, b);
  __syncthreads();
  for (int s = 128; s > 0; s >>= 1) { if (t < s) red[t] = fmaxf(red[t], red[t + s]); __syncthreads(); }
  float mx = red[0];
  __syncthreads();
  float e0 = __expf(a - mx), e1 = __expf(b - mx);
  red[t] = e0 + e1;
  __syncthreads();
  for (int s = 128; s > 0; s >>= 1) { if (t < s) red[t] += red[t + s]; __syncthreads(); }
  float inv = 1.0f / red[0];
  out[base + t]       = (__bf16)(e0 * inv);
  out[base + t + 256] = (__bf16)(e1 * inv);
}

// ---------------------------------------------------------------------------
// x = LayerNorm(delta + x); writes f32 master copy and bf16 GEMM copy
// ---------------------------------------------------------------------------
__global__ __launch_bounds__(256)
void enc_add_ln(const float* __restrict__ delta, float* __restrict__ x,
                __bf16* __restrict__ xb,
                const float* __restrict__ gamma, const float* __restrict__ beta)
{
  size_t base = (size_t)blockIdx.x * kE;
  int t = threadIdx.x;
  float v[4], s = 0.0f, s2 = 0.0f;
#pragma unroll
  for (int i = 0; i < 4; ++i) {
    int e = t + i * 256;
    float val = delta[base + e] + x[base + e];
    v[i] = val; s += val; s2 += val * val;
  }
  __shared__ float rs[256], rs2[256];
  rs[t] = s; rs2[t] = s2;
  __syncthreads();
  for (int k = 128; k > 0; k >>= 1) {
    if (t < k) { rs[t] += rs[t + k]; rs2[t] += rs2[t + k]; }
    __syncthreads();
  }
  float mu  = rs[0] * (1.0f / kE);
  float var = rs2[0] * (1.0f / kE) - mu * mu;
  float inv = rsqrtf(var + 1e-5f);
#pragma unroll
  for (int i = 0; i < 4; ++i) {
    int e = t + i * 256;
    float o = (v[i] - mu) * inv * gamma[e] + beta[e];
    x[base + e]  = o;
    xb[base + e] = (__bf16)o;
  }
}

// ---------------------------------------------------------------------------
// Final projection: out[b, :] = x[b, S*E] @ wout + bout   (K = 524288, N = 10)
// ---------------------------------------------------------------------------
__global__ __launch_bounds__(256)
void enc_final(const float* __restrict__ x, const float* __restrict__ wout,
               const float* __restrict__ bout, float* __restrict__ out)
{
  int b = blockIdx.x, t = threadIdx.x;
  const float* xr = x + (size_t)b * (kS * kE);
  float acc[kOUT];
#pragma unroll
  for (int o = 0; o < kOUT; ++o) acc[o] = 0.0f;
  for (int k = t; k < kS * kE; k += 256) {
    float xv = xr[k];
    const float* wr = wout + (size_t)k * kOUT;
#pragma unroll
    for (int o = 0; o < kOUT; ++o) acc[o] += xv * wr[o];
  }
  __shared__ float red[256];
  for (int o = 0; o < kOUT; ++o) {
    red[t] = acc[o];
    __syncthreads();
    for (int s = 128; s > 0; s >>= 1) { if (t < s) red[t] += red[t + s]; __syncthreads(); }
    if (t == 0) out[b * kOUT + o] = red[0] + bout[o];
    __syncthreads();
  }
}

// ---------------------------------------------------------------------------
// Host orchestration
// ---------------------------------------------------------------------------
extern "C" void kernel_launch(void* const* d_in, const int* in_sizes, int n_in,
                              void* d_out, int out_size, void* d_ws, size_t ws_size,
                              hipStream_t stream)
{
  const int*   tok   = (const int*)d_in[0];
  const float* emb   = (const float*)d_in[1];
  const float* wq    = (const float*)d_in[2];
  const float* bq    = (const float*)d_in[3];
  const float* wk    = (const float*)d_in[4];
  const float* bk    = (const float*)d_in[5];
  const float* wv    = (const float*)d_in[6];
  const float* bv    = (const float*)d_in[7];
  const float* wfc   = (const float*)d_in[8];
  const float* bfc   = (const float*)d_in[9];
  const float* gamma = (const float*)d_in[10];
  const float* beta  = (const float*)d_in[11];
  const float* wout  = (const float*)d_in[12];
  const float* bout  = (const float*)d_in[13];
  float* out = (float*)d_out;

  char* ws = (char*)d_ws;
  size_t off = 0;
  auto alloc = [&](size_t bytes) -> void* {
    void* p = ws + off;
    off += (bytes + 255) & ~(size_t)255;
    return p;
  };
  const size_t NTOK = (size_t)kB * kS;                 // 8192
  float*  x_f32 = (float*) alloc(NTOK * kE * 4);       // 32 MB master activations
  __bf16* x_bf  = (__bf16*)alloc(NTOK * kE * 2);       // 16 MB
  __bf16* q_bf  = (__bf16*)alloc(NTOK * kE * 2);       // 16 MB
  __bf16* k_bf  = (__bf16*)alloc(NTOK * kE * 2);       // 16 MB
  __bf16* vT_bf = (__bf16*)alloc(NTOK * kE * 2);       // 16 MB [B,H,D,S]
  float*  y_f32 = (float*) alloc(NTOK * kE * 4);       // 32 MB attn-out / fc-out
  float*  att_f = (float*) alloc((size_t)kH * kS * kS * 4);  // 16 MB per-batch scores
  __bf16* att_b = (__bf16*)alloc((size_t)kH * kS * kS * 2);  //  8 MB per-batch probs
  __bf16* wqT   = (__bf16*)alloc((size_t)kE * kE * 2);
  __bf16* wkT   = (__bf16*)alloc((size_t)kE * kE * 2);
  __bf16* wvT   = (__bf16*)alloc((size_t)kE * kE * 2);
  __bf16* wfcT  = (__bf16*)alloc((size_t)kE * kE * 2);

  dim3 blk(256);
  const int wElems = (kE * kE) / 256;
  enc_transpose_w<<<wElems, blk, 0, stream>>>(wq,  wqT);
  enc_transpose_w<<<wElems, blk, 0, stream>>>(wk,  wkT);
  enc_transpose_w<<<wElems, blk, 0, stream>>>(wv,  wvT);
  enc_transpose_w<<<wElems, blk, 0, stream>>>(wfc, wfcT);

  enc_embed<<<(int)NTOK, blk, 0, stream>>>(tok, emb, x_f32, x_bf);

  dim3 gProj(kE / 128, (int)NTOK / 128, 1);
  for (int layer = 0; layer < kNL; ++layer) {
    // Q, K (bf16) and V (bf16, transposed-per-head) projections
    enc_gemm<128, 1><<<gProj, blk, 0, stream>>>(x_bf, wqT, bq, nullptr, q_bf,
        kE, kE, kE, kE, 0, 0, 0, 1.0f);
    enc_gemm<128, 1><<<gProj, blk, 0, stream>>>(x_bf, wkT, bk, nullptr, k_bf,
        kE, kE, kE, kE, 0, 0, 0, 1.0f);
    enc_gemm<128, 2><<<gProj, blk, 0, stream>>>(x_bf, wvT, bv, nullptr, vT_bf,
        kE, kE, kE, kE, 0, 0, 0, 1.0f);

    for (int b = 0; b < kB; ++b) {
      const __bf16* qb = q_bf + (size_t)b * kS * kE;
      const __bf16* kb = k_bf + (size_t)b * kS * kE;
      // scores[h] = (q_h @ k_h^T) / sqrt(D)   (z = h, per-head offset h*D)
      dim3 gScore(kS / 128, kS / 128, kH);
      enc_gemm<128, 0><<<gScore, blk, 0, stream>>>(qb, kb, nullptr, att_f, nullptr,
          kD, kE, kE, kS,
          (long long)kD, (long long)kD, (long long)kS * kS, 0.125f);
      enc_softmax<<<kH * kS, blk, 0, stream>>>(att_f, att_b);
      // o[h] = att_h @ v_h  ->  y[b, s, h*D + d]
      dim3 gO(1, kS / 128, kH);
      enc_gemm<64, 0><<<gO, blk, 0, stream>>>(att_b,
          vT_bf + (size_t)b * kH * kD * kS, nullptr,
          y_f32 + (size_t)b * kS * kE, nullptr,
          kS, kS, kS, kE,
          (long long)kS * kS, (long long)kD * kS, (long long)kD, 1.0f);
    }
    // x = LN(attn_out + x)
    enc_add_ln<<<(int)NTOK, blk, 0, stream>>>(y_f32, x_f32, x_bf, gamma, beta);
    // fc = x @ wfc + bfc ; x = LN(fc + x)
    enc_gemm<128, 0><<<gProj, blk, 0, stream>>>(x_bf, wfcT, bfc, y_f32, nullptr,
        kE, kE, kE, kE, 0, 0, 0, 1.0f);
    enc_add_ln<<<(int)NTOK, blk, 0, stream>>>(y_f32, x_f32, x_bf, gamma, beta);
  }

  enc_final<<<kB, blk, 0, stream>>>(x_f32, wout, bout, out);
}
